// BitLinear_41506563949149
// MI455X (gfx1250) — 8-bit matrix operand order repaired and checked by reading, compiled, not run on hardware
//
#include <hip/hip_runtime.h>

typedef int v8i __attribute__((ext_vector_type(8)));
typedef int v4i __attribute__((ext_vector_type(4)));

#define K_DIM 4096
#define M_DIM 11008
#define B_DIM 16
#define NUM_GROUPS 4
#define ROWS_PER_GROUP (M_DIM / NUM_GROUPS)
#define KSTEPS (K_DIM / 64)
#define QSWZ_BYTES (B_DIM * K_DIM)
#define PFD 8

__global__ __launch_bounds__(256) void bitlinear_quant_kernel(
    const float* __restrict__ inp,
    unsigned char* __restrict__ qswz,
    float* __restrict__ invs,
    int* __restrict__ rowsum) {
  __shared__ float redf[256];
  __shared__ int redi[256];
  const int m = blockIdx.x;
  const int t = threadIdx.x;
  const float* row = inp + m * K_DIM;

  float mx = 0.0f;
  for (int i = t; i < K_DIM; i += 256) mx = fmaxf(mx, fabsf(row[i]));
  redf[t] = mx;
  __syncthreads();
  for (int s = 128; s > 0; s >>= 1) {
    if (t < s) redf[t] = fmaxf(redf[t], redf[t + s]);
    __syncthreads();
  }
  const float maxabs = fmaxf(redf[0], 1e-5f);
  const float s127 = 127.0f / maxabs;

  int mysum = 0;
  for (int i = t; i < K_DIM; i += 256) {
    float v = rintf(row[i] * s127);
    v = fminf(fmaxf(v, -128.0f), 127.0f);
    const int q = (int)v;
    mysum += q;
    const int kt = i >> 6;
    const int r = i & 63;
    const int b = r & 3;
    const int j = ((r >> 2) & 3) | (((r >> 5) & 1) << 2);
    const int hi = (r >> 4) & 1;
    const int lane = m + hi * 16;
    qswz[kt * 1024 + lane * 32 + j * 4 + b] = (unsigned char)(q & 0xFF);
  }
  redi[t] = mysum;
  __syncthreads();
  for (int s = 128; s > 0; s >>= 1) {
    if (t < s) redi[t] += redi[t + s];
    __syncthreads();
  }
  if (t == 0) {
    invs[m] = maxabs / 127.0f;
    rowsum[m] = redi[0];
  }
}

__device__ __forceinline__ int unpack4u(unsigned p) {
  return (int)((p & 0x3u) | ((p << 6) & 0x300u) | ((p << 12) & 0x30000u) |
               ((p << 18) & 0x3000000u));
}

__device__ __forceinline__ v8i lds_unpack(const int* lut, v4i plo, v4i phi) {
  v8i b;
  b[0] = lut[plo[0]]; b[1] = lut[plo[1]]; b[2] = lut[plo[2]]; b[3] = lut[plo[3]];
  b[4] = lut[phi[0]]; b[5] = lut[phi[1]]; b[6] = lut[phi[2]]; b[7] = lut[phi[3]];
  return b;
}

__global__ __launch_bounds__(32) void bitlinear_gemm_kernel(
    const unsigned char* __restrict__ qswz,
    const int* __restrict__ wpacked,
    const float* __restrict__ wscale,
    const float* __restrict__ invs,
    const int* __restrict__ rowsum,
    float* __restrict__ out) {
  __shared__ int lut[256];
  const int lane = threadIdx.x;
#pragma unroll
  for (int e = lane; e < 256; e += 32) lut[e] = unpack4u((unsigned)e);
  __syncthreads();

  const int tile0 = blockIdx.x * 2;
  const int n0 = lane & 15;
  const int half = lane >> 4;
  const int col0 = tile0 * 16 + n0;
  const int col1 = col0 + 16;
  const int* wrow0 = wpacked + (long)col0 * (K_DIM / 4);
  const int* wrow1 = wpacked + (long)col1 * (K_DIM / 4);
  const unsigned char* abase = qswz + lane * 32;
  const int ebOff = half * 4;

  v8i acc0; v8i acc1;
#pragma unroll
  for (int r = 0; r < 8; ++r) { acc0[r] = 0; acc1[r] = 0; }

  v4i alo[2], ahi[2], q0[2], q1[2], q2[2], q3[2];

#define LOAD_FRAG(S, KT)                                    \
  do {                                                      \
    const int kts_ = (KT);                                  \
    alo[S] = *(const v4i*)(abase + kts_ * 1024);            \
    ahi[S] = *(const v4i*)(abase + kts_ * 1024 + 16);       \
    const int eb_ = kts_ * 16 + ebOff;                      \
    q0[S] = *(const v4i*)(wrow0 + eb_);                     \
    q1[S] = *(const v4i*)(wrow0 + eb_ + 8);                 \
    q2[S] = *(const v4i*)(wrow1 + eb_);                     \
    q3[S] = *(const v4i*)(wrow1 + eb_ + 8);                 \
  } while (0)

#define CONSUME(S)                                                            \
  do {                                                                        \
    const v8i a_ = __builtin_shufflevector(alo[S], ahi[S], 0,1,2,3,4,5,6,7);  \
    const v8i b0_ = lds_unpack(lut, q0[S], q1[S]);                            \
    const v8i b1_ = lds_unpack(lut, q2[S], q3[S]);                            \
    acc0 = __builtin_amdgcn_wmma_i32_16x16x64_iu8(true, a_, false, b0_, acc0, \
                                                  false, false);              \
    acc1 = __builtin_amdgcn_wmma_i32_16x16x64_iu8(true, a_, false, b1_, acc1, \
                                                  false, false);              \
  } while (0)

  LOAD_FRAG(0, 0);
  for (int kt = 0; kt < KSTEPS; kt += 2) {
    LOAD_FRAG(1, kt + 1);
    {
      int ktp = kt + PFD; if (ktp > KSTEPS - 1) ktp = KSTEPS - 1;
      const int ebp = ktp * 16 + ebOff;
      __builtin_prefetch((const void*)(wrow0 + ebp), 0, 3);
      __builtin_prefetch((const void*)(wrow1 + ebp), 0, 3);
    }
    CONSUME(0);
    int ktn = kt + 2; if (ktn > KSTEPS - 2) ktn = KSTEPS - 2;
    LOAD_FRAG(0, ktn);
    CONSUME(1);
  }
#undef LOAD_FRAG
#undef CONSUME

  const float g0 = wscale[(tile0 * 16) / ROWS_PER_GROUP];
  const float g1 = wscale[((tile0 + 1) * 16) / ROWS_PER_GROUP];
  const int hi8 = half * 8;
#pragma unroll
  for (int r = 0; r < 8; ++r) {
    const int m = r + hi8;
    const float fi = invs[m];
    const int rs = rowsum[m];
    out[(long)m * M_DIM + col0] = (float)(acc0[r] - rs) * fi * g0;
    out[(long)m * M_DIM + col1] = (float)(acc1[r] - rs) * fi * g1;
  }
}

extern "C" void kernel_launch(void* const* d_in, const int* in_sizes, int n_in,
                              void* d_out, int out_size, void* d_ws,
                              size_t ws_size, hipStream_t stream) {
  const float* inp = (const float*)d_in[0];
  const int* wpacked = (const int*)d_in[1];
  const float* wscale = (const float*)d_in[2];
  float* out = (float*)d_out;

  unsigned char* qswz = (unsigned char*)d_ws;
  float* invs = (float*)((char*)d_ws + QSWZ_BYTES);
  int* rowsum = (int*)((char*)d_ws + QSWZ_BYTES + 64);

  bitlinear_quant_kernel<<<16, 256, 0, stream>>>(inp, qswz, invs, rowsum);
  bitlinear_gemm_kernel<<<344, 32, 0, stream>>>(qswz, wpacked, wscale, invs,
                                                rowsum, out);
}
